// HeteroGNS_23527830848232
// MI455X (gfx1250) — matrix operand order repaired and checked by reading, not compiled, not run on hardware
//
#include <hip/hip_runtime.h>
#include <math.h>

// ---------------- constants from the reference ----------------
#define LATENT 128
#define BDIM   128
#define NB     128
#define DFF    512
#define HEADS  2
#define HD     64           // BDIM / HEADS
#define NN     5000
#define E1SZ   40000
#define E2SZ   20000
#define FIN    16
#define FEDGE  8
#define FOUT   3
#define STEPS  10
#define TLAYERS 3

typedef __attribute__((ext_vector_type(16))) __bf16 v16bf;
typedef __attribute__((ext_vector_type(8)))  float  v8f;

__device__ __forceinline__ v8f wmma_bf16(v16bf a, v16bf b, v8f c) {
    return __builtin_amdgcn_wmma_f32_16x16x32_bf16(false, a, false, b, (short)0, c, false, false);
}

template<int B>
__device__ __forceinline__ void pack8(v16bf& a, float4 u, float4 v) {
    a[B + 0] = (__bf16)u.x; a[B + 1] = (__bf16)u.y; a[B + 2] = (__bf16)u.z; a[B + 3] = (__bf16)u.w;
    a[B + 4] = (__bf16)v.x; a[B + 5] = (__bf16)v.y; a[B + 6] = (__bf16)v.z; a[B + 7] = (__bf16)v.w;
}

// =====================================================================
// WMMA GEMM:  Y[M,N] = act( alpha * X[M,K] @ Wop[K,N] + bias )
//   TRANSB=0: Wop[k][n] = W[k*ldw + n];  TRANSB=1: Wop[k][n] = W[n*ldw + k]
//   VEC=1: A rows (and TRANSB B rows) loaded as float4 (requires ldx%4==0)
//   bf16 operands, f32 accumulate (v_wmma_f32_16x16x32_bf16).
//   4 waves / block; each wave owns a 32x32 output tile (2x2 WMMA tiles):
//   one A/B fragment pair feeds 4 WMMAs. No branches in the K loop.
// =====================================================================
template<int ACT, int TRANSB, int VEC>
__global__ __launch_bounds__(128)
void wmma_gemm_kernel(const float* __restrict__ X, const float* __restrict__ W,
                      const float* __restrict__ bias, float* __restrict__ Y,
                      int M, int K, int N, int ldx, int ldw, int ldy, float alpha)
{
    const int wave = threadIdx.x >> 5;
    const int lane = threadIdx.x & 31;
    const int hf   = lane >> 4;      // 0: lanes 0-15, 1: lanes 16-31
    const int l16  = lane & 15;

    const int n0 = blockIdx.x * 32;
    const int m0 = (blockIdx.y * 4 + wave) * 32;
    if (m0 >= M) return;             // wave-uniform: EXEC stays all-ones for WMMA

    // clamped A rows (ragged M handled by clamping loads + guarding stores)
    int ar0 = m0 + l16;      if (ar0 >= M) ar0 = M - 1;
    int ar1 = m0 + 16 + l16; if (ar1 >= M) ar1 = M - 1;
    const float* __restrict__ xr0 = X + (size_t)ar0 * ldx;
    const float* __restrict__ xr1 = X + (size_t)ar1 * ldx;

    // clamped B columns (cols >= N produce garbage accumulator columns that
    // are never stored -- no guards needed inside the K loop)
    const int c0 = n0 + l16, c1 = n0 + 16 + l16;
    const int c0c = (c0 < N) ? c0 : (N - 1);
    const int c1c = (c1 < N) ? c1 : (N - 1);
    const float* __restrict__ wr0 = TRANSB ? (W + (size_t)c0c * ldw) : (W + c0c);
    const float* __restrict__ wr1 = TRANSB ? (W + (size_t)c1c * ldw) : (W + c1c);

    v8f acc00 = {}, acc01 = {}, acc10 = {}, acc11 = {};

    const int kfull = K & ~31;
    for (int k0 = 0; k0 < kfull; k0 += 32) {
        v16bf a0, a1, b0, b1;
        // ---- A fragments (16-bit A 16x32 layout: per lane two contiguous
        //      8-float runs: [k0+hf*8, +8) and [k0+16+hf*8, +8)) ----
        if (VEC) {
            const float* p0 = xr0 + k0 + hf * 8;
            const float* p1 = xr1 + k0 + hf * 8;
            __builtin_prefetch(p0 + 128, 0, 3);
            pack8<0>(a0, *(const float4*)(p0),      *(const float4*)(p0 + 4));
            pack8<8>(a0, *(const float4*)(p0 + 16), *(const float4*)(p0 + 20));
            pack8<0>(a1, *(const float4*)(p1),      *(const float4*)(p1 + 4));
            pack8<8>(a1, *(const float4*)(p1 + 16), *(const float4*)(p1 + 20));
        } else {
#pragma unroll
            for (int j = 0; j < 16; ++j) {
                int kk = k0 + ((j < 8) ? j : j + 8) + hf * 8;
                a0[j] = (__bf16)xr0[kk];
                a1[j] = (__bf16)xr1[kk];
            }
        }
        // ---- B fragments (16-bit B 32x16: elem j -> K = k0 + (j<8 ? j : j+8) + hf*8) ----
        if (TRANSB) {   // k-contiguous rows: vector loads
            const float* q0 = wr0 + k0 + hf * 8;
            const float* q1 = wr1 + k0 + hf * 8;
            pack8<0>(b0, *(const float4*)(q0),     *(const float4*)(q0 + 4));
            pack8<8>(b0, *(const float4*)(q0 + 16), *(const float4*)(q0 + 20));
            pack8<0>(b1, *(const float4*)(q1),     *(const float4*)(q1 + 4));
            pack8<8>(b1, *(const float4*)(q1 + 16), *(const float4*)(q1 + 20));
        } else {        // column-strided, unconditional scalar loads
#pragma unroll
            for (int j = 0; j < 16; ++j) {
                size_t off = (size_t)(k0 + ((j < 8) ? j : j + 8) + hf * 8) * ldw;
                b0[j] = (__bf16)wr0[off];
                b1[j] = (__bf16)wr1[off];
            }
        }
        acc00 = wmma_bf16(a0, b0, acc00);
        acc01 = wmma_bf16(a0, b1, acc01);
        acc10 = wmma_bf16(a1, b0, acc10);
        acc11 = wmma_bf16(a1, b1, acc11);
    }

    if (kfull < K) {   // ragged-K tail: clamped loads + value masking (v_cndmask, no branches)
        v16bf a0, a1, b0, b1;
#pragma unroll
        for (int j = 0; j < 16; ++j) {
            int kk = kfull + ((j < 8) ? j : j + 8) + hf * 8;
            int kc = (kk < K) ? kk : (K - 1);
            float v0 = xr0[kc], v1 = xr1[kc];
            a0[j] = (__bf16)((kk < K) ? v0 : 0.0f);
            a1[j] = (__bf16)((kk < K) ? v1 : 0.0f);
            int kb  = kfull + ((j < 8) ? j : j + 8) + hf * 8;
            int kbc = (kb < K) ? kb : (K - 1);
            float w0 = TRANSB ? wr0[kbc] : wr0[(size_t)kbc * ldw];
            float w1 = TRANSB ? wr1[kbc] : wr1[(size_t)kbc * ldw];
            b0[j] = (__bf16)((kb < K) ? w0 : 0.0f);
            b1[j] = (__bf16)((kb < K) ? w1 : 0.0f);
        }
        acc00 = wmma_bf16(a0, b0, acc00);
        acc01 = wmma_bf16(a0, b1, acc01);
        acc10 = wmma_bf16(a1, b0, acc10);
        acc11 = wmma_bf16(a1, b1, acc11);
    }

    // ---- epilogue: bias + act, guarded stores ----
    float bv0 = bias ? bias[c0c] : 0.0f;   // clamped; only consumed when c<N
    float bv1 = bias ? bias[c1c] : 0.0f;
#pragma unroll
    for (int r = 0; r < 8; ++r) {          // C/D: VGPR r -> row base + hf*8 + r
        int row0 = m0 + hf * 8 + r;
        int row1 = row0 + 16;
        if (row0 < M) {
            if (c0 < N) { float v = acc00[r] * alpha + bv0; if (ACT) v = fmaxf(v, 0.f); Y[(size_t)row0 * ldy + c0] = v; }
            if (c1 < N) { float v = acc01[r] * alpha + bv1; if (ACT) v = fmaxf(v, 0.f); Y[(size_t)row0 * ldy + c1] = v; }
        }
        if (row1 < M) {
            if (c0 < N) { float v = acc10[r] * alpha + bv0; if (ACT) v = fmaxf(v, 0.f); Y[(size_t)row1 * ldy + c0] = v; }
            if (c1 < N) { float v = acc11[r] * alpha + bv1; if (ACT) v = fmaxf(v, 0.f); Y[(size_t)row1 * ldy + c1] = v; }
        }
    }
}

// ---------------- LayerNorm (one wave32 per row, optional fused residual) ----
__global__ __launch_bounds__(128)
void layernorm_kernel(const float* __restrict__ x, const float* __restrict__ res,
                      const float* __restrict__ g, const float* __restrict__ b,
                      float* __restrict__ y, int M, int D)
{
    int row = blockIdx.x * (blockDim.x >> 5) + (threadIdx.x >> 5);
    if (row >= M) return;
    int lane = threadIdx.x & 31;
    const float* xr = x + (size_t)row * D;
    const float* rr = res ? res + (size_t)row * D : nullptr;
    float s = 0.f, s2 = 0.f;
    for (int c = lane; c < D; c += 32) {
        float v = xr[c] + (rr ? rr[c] : 0.f);
        s += v; s2 += v * v;
    }
    for (int o = 16; o; o >>= 1) { s += __shfl_xor(s, o, 32); s2 += __shfl_xor(s2, o, 32); }
    float mean = s / D;
    float inv  = rsqrtf(s2 / D - mean * mean + 1e-5f);
    float* yr = y + (size_t)row * D;
    for (int c = lane; c < D; c += 32) {
        float v = xr[c] + (rr ? rr[c] : 0.f);
        yr[c] = (v - mean) * inv * g[c] + b[c];
    }
}

// ---------------- softmax over rows (wave32 per row) -------------------------
__global__ __launch_bounds__(128)
void softmax_rows_kernel(float* __restrict__ a, int rows, int cols)
{
    int row = blockIdx.x * (blockDim.x >> 5) + (threadIdx.x >> 5);
    if (row >= rows) return;
    int lane = threadIdx.x & 31;
    float* p = a + (size_t)row * cols;
    float mx = -1e30f;
    for (int c = lane; c < cols; c += 32) mx = fmaxf(mx, p[c]);
    for (int o = 16; o; o >>= 1) mx = fmaxf(mx, __shfl_xor(mx, o, 32));
    float s = 0.f;
    for (int c = lane; c < cols; c += 32) { float e = __expf(p[c] - mx); p[c] = e; s += e; }
    for (int o = 16; o; o >>= 1) s += __shfl_xor(s, o, 32);
    float inv = 1.0f / s;
    for (int c = lane; c < cols; c += 32) p[c] *= inv;
}

// ---------------- edge feature concat: [x[dst], x[src], ef] -> [E,384] -------
__global__ void edge_concat_kernel(const float* __restrict__ x, const float* __restrict__ ef,
                                   const int* __restrict__ src, const int* __restrict__ dst,
                                   float* __restrict__ out, int E)
{
    int i = blockIdx.x * blockDim.x + threadIdx.x;
    int total = E * 384;
    if (i >= total) return;
    int e = i / 384, c = i % 384;
    float v;
    if (c < 128)       v = x[(size_t)dst[e] * 128 + c];
    else if (c < 256)  v = x[(size_t)src[e] * 128 + (c - 128)];
    else               v = ef[(size_t)e * 128 + (c - 256)];
    out[i] = v;
}

// ---------------- node concat: [agg, x, bvec] -> [N,384] ---------------------
__global__ void node_concat_kernel(const float* __restrict__ agg, const float* __restrict__ x,
                                   const float* __restrict__ bvec, float* __restrict__ out, int n)
{
    int i = blockIdx.x * blockDim.x + threadIdx.x;
    int total = n * 384;
    if (i >= total) return;
    int r = i / 384, c = i % 384;
    float v;
    if (c < 128)       v = agg[(size_t)r * 128 + c];
    else if (c < 256)  v = x[(size_t)r * 128 + (c - 128)];
    else               v = bvec[c - 256];
    out[i] = v;
}

// ---------------- misc elementwise ------------------------------------------
__global__ void zero_kernel(float* __restrict__ p, int n)
{
    int i = blockIdx.x * blockDim.x + threadIdx.x;
    if (i < n) p[i] = 0.0f;
}
__global__ void scatter_add_kernel(const float* __restrict__ msg, const int* __restrict__ dst,
                                   float* __restrict__ agg, float* __restrict__ cnt, int E)
{
    int i = blockIdx.x * blockDim.x + threadIdx.x;
    int total = E * 128;
    if (i >= total) return;
    int e = i >> 7, c = i & 127;
    atomicAdd(&agg[(size_t)dst[e] * 128 + c], msg[i]);
    if (c == 0) atomicAdd(&cnt[dst[e]], 1.0f);
}
__global__ void div_cnt_kernel(float* __restrict__ agg, const float* __restrict__ cnt, int n)
{
    int i = blockIdx.x * blockDim.x + threadIdx.x;
    if (i < n * 128) agg[i] /= fmaxf(cnt[i >> 7], 1.0f);
}
__global__ void add_inplace_kernel(float* __restrict__ y, const float* __restrict__ a, int n)
{
    int i = blockIdx.x * blockDim.x + threadIdx.x;
    if (i < n) y[i] += a[i];
}
__global__ void add2_kernel(const float* __restrict__ a, const float* __restrict__ b,
                            float* __restrict__ out, int n)
{
    int i = blockIdx.x * blockDim.x + threadIdx.x;
    if (i < n) out[i] = a[i] + b[i];
}
__global__ void colmean_kernel(const float* __restrict__ x, float* __restrict__ out,
                               int rows, int cols)
{
    int c = blockIdx.x * blockDim.x + threadIdx.x;
    if (c >= cols) return;
    float s = 0.f;
    for (int r = 0; r < rows; ++r) s += x[(size_t)r * cols + c];
    out[c] = s / rows;
}

// =====================================================================
// Host-side orchestration
// =====================================================================
namespace {

struct Dense { const float* w; const float* b; };
struct LNP   { const float* g; const float* b; };
struct TLayer { Dense ff1, ff2; LNP ln1, ln2; Dense wk, wo, wq, wv; };
struct Trans  { Dense embed; TLayer layer[TLAYERS]; };
struct Step   { LNP edge_ln; Dense edge_mlp[3]; LNP node_ln; Dense node_mlp[3]; Trans trans; };

// JAX tree_flatten: dict keys alphabetical; Dense {"b","w"} -> b,w; LN {"b","g"} -> b,g
struct Cur {
    void* const* p; int i;
    const float* next() { return (const float*)p[i++]; }
};
static Dense rdDense(Cur& c) { Dense d; d.b = c.next(); d.w = c.next(); return d; }
static LNP   rdLN(Cur& c)    { LNP l; l.b = c.next(); l.g = c.next(); return l; }
static void  rdMLP(Cur& c, Dense* m) { for (int i = 0; i < 3; ++i) m[i] = rdDense(c); }
static TLayer rdTL(Cur& c) {
    TLayer t;
    t.ff1 = rdDense(c); t.ff2 = rdDense(c);
    t.ln1 = rdLN(c);    t.ln2 = rdLN(c);
    t.wk = rdDense(c); t.wo = rdDense(c); t.wq = rdDense(c); t.wv = rdDense(c);
    return t;
}
static Trans rdTrans(Cur& c) {
    Trans t; t.embed = rdDense(c);
    for (int i = 0; i < TLAYERS; ++i) t.layer[i] = rdTL(c);
    return t;
}
static Step rdStep(Cur& c) {
    Step s;
    s.edge_ln = rdLN(c); rdMLP(c, s.edge_mlp);
    s.node_ln = rdLN(c); rdMLP(c, s.node_mlp);
    s.trans = rdTrans(c);
    return s;
}

static void gemm(hipStream_t st, const float* X, const float* W, const float* bias,
                 float* Y, int M, int K, int N, int ldx, int ldw, int ldy,
                 int act, int transB, float alpha)
{
    dim3 grid((N + 31) / 32, (M + 127) / 128), block(128);
    const bool vec = (ldx % 4) == 0;
    if (transB) {   // attention scores: always vectorizable, no activation
        wmma_gemm_kernel<0,1,1><<<grid, block, 0, st>>>(X, W, bias, Y, M, K, N, ldx, ldw, ldy, alpha);
    } else if (vec) {
        if (act) wmma_gemm_kernel<1,0,1><<<grid, block, 0, st>>>(X, W, bias, Y, M, K, N, ldx, ldw, ldy, alpha);
        else     wmma_gemm_kernel<0,0,1><<<grid, block, 0, st>>>(X, W, bias, Y, M, K, N, ldx, ldw, ldy, alpha);
    } else {
        if (act) wmma_gemm_kernel<1,0,0><<<grid, block, 0, st>>>(X, W, bias, Y, M, K, N, ldx, ldw, ldy, alpha);
        else     wmma_gemm_kernel<0,0,0><<<grid, block, 0, st>>>(X, W, bias, Y, M, K, N, ldx, ldw, ldy, alpha);
    }
}

static void layernorm(hipStream_t st, const float* x, const float* res, LNP ln,
                      float* y, int M, int D)
{
    dim3 grid((M + 3) / 4), block(128);   // 4 wave32 rows / block
    layernorm_kernel<<<grid, block, 0, st>>>(x, res, ln.g, ln.b, y, M, D);
}

// 3-layer MLP: ReLU, ReLU, Identity (build_mlp defaults)
static void mlp3(hipStream_t st, const float* X, const Dense* m, int M, int K0, int doutLast,
                 float* t1, float* t2, float* out)
{
    gemm(st, X,  m[0].w, m[0].b, t1,  M, K0,  128,      K0,  128,      128,      1, 0, 1.f);
    gemm(st, t1, m[1].w, m[1].b, t2,  M, 128, 128,      128, 128,      128,      1, 0, 1.f);
    gemm(st, t2, m[2].w, m[2].b, out, M, 128, doutLast, 128, doutLast, doutLast, 0, 0, 1.f);
}

struct TransWS { float *tx, *tq, *tk, *tv, *to, *ttmp, *tff, *att, *bvec; };

// boundary transformer encoder -> pooled bvec[BDIM]
static void transformer(hipStream_t st, const Trans& T, const float* bnd, TransWS& w)
{
    const float scale = 1.0f / sqrtf((float)HD);
    gemm(st, bnd, T.embed.w, T.embed.b, w.tx, NB, 3, BDIM, 3, BDIM, BDIM, 0, 0, 1.f);
    for (int L = 0; L < TLAYERS; ++L) {
        const TLayer& ly = T.layer[L];
        gemm(st, w.tx, ly.wq.w, ly.wq.b, w.tq, NB, BDIM, BDIM, BDIM, BDIM, BDIM, 0, 0, 1.f);
        gemm(st, w.tx, ly.wk.w, ly.wk.b, w.tk, NB, BDIM, BDIM, BDIM, BDIM, BDIM, 0, 0, 1.f);
        gemm(st, w.tx, ly.wv.w, ly.wv.b, w.tv, NB, BDIM, BDIM, BDIM, BDIM, BDIM, 0, 0, 1.f);
        for (int h = 0; h < HEADS; ++h)  // scores = scale * q_h @ k_h^T  (WMMA, B transposed)
            gemm(st, w.tq + h * HD, w.tk + h * HD, nullptr, w.att + h * NB * NB,
                 NB, HD, NB, BDIM, BDIM, NB, 0, 1, scale);
        softmax_rows_kernel<<<dim3((HEADS * NB + 3) / 4), 128, 0, st>>>(w.att, HEADS * NB, NB);
        for (int h = 0; h < HEADS; ++h)  // o_h = att_h @ v_h (submatrix strides)
            gemm(st, w.att + h * NB * NB, w.tv + h * HD, nullptr, w.to + h * HD,
                 NB, NB, HD, NB, BDIM, BDIM, 0, 0, 1.f);
        gemm(st, w.to, ly.wo.w, ly.wo.b, w.ttmp, NB, BDIM, BDIM, BDIM, BDIM, BDIM, 0, 0, 1.f);
        layernorm(st, w.ttmp, w.tx, ly.ln1, w.tx, NB, BDIM);
        gemm(st, w.tx, ly.ff1.w, ly.ff1.b, w.tff, NB, BDIM, DFF, BDIM, DFF, DFF, 1, 0, 1.f);
        gemm(st, w.tff, ly.ff2.w, ly.ff2.b, w.ttmp, NB, DFF, BDIM, DFF, BDIM, BDIM, 0, 0, 1.f);
        layernorm(st, w.ttmp, w.tx, ly.ln2, w.tx, NB, BDIM);
    }
    colmean_kernel<<<1, 128, 0, st>>>(w.tx, w.bvec, NB, BDIM);
}

struct StepWS {
    float *ecat, *eh1, *eh2, *msg;
    float *agg, *cnt;
    float *ncat, *nh1, *nh2, *xu;
};

static void interaction(hipStream_t st, const Step& S, float* x, float* ef,
                        const int* src, const int* dst, int E, const float* bnd,
                        TransWS& tw, StepWS& sw)
{
    transformer(st, S.trans, bnd, tw);
    // edge update: msg = LN(edge_mlp([x[dst], x[src], ef]))
    {
        int total = E * 384;
        edge_concat_kernel<<<(total + 255) / 256, 256, 0, st>>>(x, ef, src, dst, sw.ecat, E);
    }
    mlp3(st, sw.ecat, S.edge_mlp, E, 384, 128, sw.eh1, sw.eh2, sw.eh1);
    layernorm(st, sw.eh1, nullptr, S.edge_ln, sw.msg, E, 128);
    // segment mean over dst
    zero_kernel<<<(NN * 128 + 255) / 256, 256, 0, st>>>(sw.agg, NN * 128);
    zero_kernel<<<(NN + 255) / 256, 256, 0, st>>>(sw.cnt, NN);
    scatter_add_kernel<<<(E * 128 + 255) / 256, 256, 0, st>>>(sw.msg, dst, sw.agg, sw.cnt, E);
    div_cnt_kernel<<<(NN * 128 + 255) / 256, 256, 0, st>>>(sw.agg, sw.cnt, NN);
    // node update: xu = LN(node_mlp([agg, x, bvec]))
    {
        int total = NN * 384;
        node_concat_kernel<<<(total + 255) / 256, 256, 0, st>>>(sw.agg, x, tw.bvec, sw.ncat, NN);
    }
    mlp3(st, sw.ncat, S.node_mlp, NN, 384, 128, sw.nh1, sw.nh2, sw.nh1);
    layernorm(st, sw.nh1, nullptr, S.node_ln, sw.xu, NN, 128);
    // residuals
    add_inplace_kernel<<<(NN * 128 + 255) / 256, 256, 0, st>>>(x, sw.xu, NN * 128);
    add_inplace_kernel<<<(E * 128 + 255) / 256, 256, 0, st>>>(ef, sw.msg, E * 128);
}

} // namespace

extern "C" void kernel_launch(void* const* d_in, const int* in_sizes, int n_in,
                              void* d_out, int out_size, void* d_ws, size_t ws_size,
                              hipStream_t stream)
{
    (void)in_sizes; (void)n_in; (void)out_size; (void)ws_size;
    // ---- inputs (setup_inputs order) ----
    const float* d_x    = (const float*)d_in[0];
    const int*   d_ei1  = (const int*)  d_in[1];     // [2, E1]
    const float* d_ef1  = (const float*)d_in[2];
    const float* d_bnd1 = (const float*)d_in[3];
    const float* d_xi   = (const float*)d_in[4];
    const int*   d_ei2  = (const int*)  d_in[5];     // [2, E2]
    const float* d_ef2  = (const float*)d_in[6];
    const float* d_bnd2 = (const float*)d_in[7];

    // ---- params: JAX tree_flatten (sorted keys: dec, enc, proc_g1, proc_g2) ----
    Cur cur{d_in, 8};
    Dense dec_mlp[3], dec_mlp_i[3];
    rdMLP(cur, dec_mlp); rdMLP(cur, dec_mlp_i);
    LNP   enc_eln  = rdLN(cur), enc_eln_i = rdLN(cur);
    Dense enc_em[3], enc_em_i[3];
    rdMLP(cur, enc_em); rdMLP(cur, enc_em_i);
    LNP   enc_nln  = rdLN(cur), enc_nln_i = rdLN(cur);
    Dense enc_nm[3], enc_nm_i[3];
    rdMLP(cur, enc_nm); rdMLP(cur, enc_nm_i);
    static Step g1[STEPS], g2[STEPS];
    for (int i = 0; i < STEPS; ++i) g1[i] = rdStep(cur);
    for (int i = 0; i < STEPS; ++i) g2[i] = rdStep(cur);

    // ---- workspace bump allocation (floats) ----
    float* wsp = (float*)d_ws;
    auto alloc = [&](size_t n) { float* p = wsp; wsp += n; return p; };
    float* xg1  = alloc((size_t)NN * 128);
    float* xg2  = alloc((size_t)NN * 128);
    float* efg1 = alloc((size_t)E1SZ * 128);
    float* efg2 = alloc((size_t)E2SZ * 128);
    StepWS sw;
    sw.ecat = alloc((size_t)E1SZ * 384);
    sw.eh1  = alloc((size_t)E1SZ * 128);
    sw.eh2  = alloc((size_t)E1SZ * 128);
    sw.msg  = alloc((size_t)E1SZ * 128);
    sw.agg  = alloc((size_t)NN * 128);
    sw.cnt  = alloc((size_t)NN);
    sw.ncat = alloc((size_t)NN * 384);
    sw.nh1  = alloc((size_t)NN * 128);
    sw.nh2  = alloc((size_t)NN * 128);
    sw.xu   = alloc((size_t)NN * 128);
    TransWS tw;
    tw.tx   = alloc(NB * BDIM);
    tw.tq   = alloc(NB * BDIM);
    tw.tk   = alloc(NB * BDIM);
    tw.tv   = alloc(NB * BDIM);
    tw.to   = alloc(NB * BDIM);
    tw.ttmp = alloc(NB * BDIM);
    tw.tff  = alloc(NB * DFF);
    tw.att  = alloc(HEADS * NB * NB);
    tw.bvec = alloc(BDIM);
    float* dtmp1 = alloc((size_t)NN * 4);
    float* dtmp2 = alloc((size_t)NN * 4);

    const int* src1 = d_ei1;              const int* dst1 = d_ei1 + E1SZ;
    const int* src2 = d_ei2;              const int* dst2 = d_ei2 + E2SZ;

    // ---- encoder ----
    mlp3(stream, d_x, enc_nm, NN, FIN, 128, sw.nh1, sw.nh2, sw.nh1);
    layernorm(stream, sw.nh1, nullptr, enc_nln, xg1, NN, 128);
    mlp3(stream, d_ef1, enc_em, E1SZ, FEDGE, 128, sw.eh1, sw.eh2, sw.eh1);
    layernorm(stream, sw.eh1, nullptr, enc_eln, efg1, E1SZ, 128);
    mlp3(stream, d_xi, enc_nm_i, NN, FIN, 128, sw.nh1, sw.nh2, sw.nh1);
    layernorm(stream, sw.nh1, nullptr, enc_nln_i, xg2, NN, 128);
    mlp3(stream, d_ef2, enc_em_i, E2SZ, FEDGE, 128, sw.eh1, sw.eh2, sw.eh1);
    layernorm(stream, sw.eh1, nullptr, enc_eln_i, efg2, E2SZ, 128);

    // ---- processors ----
    for (int s = 0; s < STEPS; ++s)
        interaction(stream, g1[s], xg1, efg1, src1, dst1, E1SZ, d_bnd1, tw, sw);
    for (int s = 0; s < STEPS; ++s)
        interaction(stream, g2[s], xg2, efg2, src2, dst2, E2SZ, d_bnd2, tw, sw);

    // ---- decoder ----
    mlp3(stream, xg1, dec_mlp,   NN, 128, FOUT, sw.nh1, sw.nh2, dtmp1);
    mlp3(stream, xg2, dec_mlp_i, NN, 128, FOUT, sw.nh1, sw.nh2, dtmp2);
    add2_kernel<<<(NN * FOUT + 255) / 256, 256, 0, stream>>>(dtmp1, dtmp2, (float*)d_out, NN * FOUT);
}
